// GATEncoder_48086453846460
// MI455X (gfx1250) — matrix operand order repaired and checked by reading, not compiled, not run on hardware
//
#include <hip/hip_runtime.h>
#include <math.h>

// ---------------------------------------------------------------------------
// Types / helpers
// ---------------------------------------------------------------------------
typedef __bf16 bf16;
typedef __attribute__((ext_vector_type(16))) __bf16 v16bf;
typedef __attribute__((ext_vector_type(8)))  __bf16 v8bf;
typedef __attribute__((ext_vector_type(8)))  float  v8f;

union BFrag { v16bf v; v8bf h[2]; };

static __device__ __forceinline__ bf16 f2bf(float f) {
  unsigned u = __float_as_uint(f);
  u += ((u >> 16) & 1u) + 0x7FFFu;          // round-to-nearest-even
  union { unsigned short s; bf16 b; } cv;
  cv.s = (unsigned short)(u >> 16);
  return cv.b;
}

// order-preserving float<->uint encoding for atomicMax over signed floats
static __device__ __forceinline__ unsigned f2ord(float f) {
  unsigned b = __float_as_uint(f);
  return (b & 0x80000000u) ? ~b : (b | 0x80000000u);
}
static __device__ __forceinline__ float ord2f(unsigned u) {
  unsigned b = (u & 0x80000000u) ? (u & 0x7FFFFFFFu) : ~u;
  return __uint_as_float(b);
}

// ---------------------------------------------------------------------------
// Elementwise conversion kernels
// ---------------------------------------------------------------------------
__global__ void k_f32_to_bf16(const float* __restrict__ x, bf16* __restrict__ y,
                              size_t total) {
  size_t i = (size_t)blockIdx.x * blockDim.x + threadIdx.x;
  if (i < total) y[i] = f2bf(x[i]);
}

// W [K,N] f32 row-major  ->  Wt [N,K] bf16 row-major (B operand, N-major)
__global__ void k_transpose_to_bf16(const float* __restrict__ W,
                                    bf16* __restrict__ Wt, int K, int N) {
  size_t i = (size_t)blockIdx.x * blockDim.x + threadIdx.x;
  size_t total = (size_t)K * N;
  if (i >= total) return;
  int k = (int)(i / N);
  int n = (int)(i % N);
  Wt[(size_t)n * K + k] = f2bf(W[i]);
}

__global__ void k_fill_f32(float* __restrict__ p, float v, size_t total) {
  size_t i = (size_t)blockIdx.x * blockDim.x + threadIdx.x;
  if (i < total) p[i] = v;
}
__global__ void k_fill_u32(unsigned* __restrict__ p, unsigned v, size_t total) {
  size_t i = (size_t)blockIdx.x * blockDim.x + threadIdx.x;
  if (i < total) p[i] = v;
}

// ---------------------------------------------------------------------------
// bf16 WMMA GEMM:  C[M,N] (f32) = A[M,K] (bf16, row-major) x Bt[N,K]^T (bf16)
// Block = 256 threads = 8 waves; block tile 128(M) x 64(N); wave tile 16x64.
// A fragment layout (16-bit A 16x32): lane<16 row M=lane holds
// K = k0..k0+15 contiguous; lane>=16 holds K = k0+16..k0+31 contiguous.
// B fragment (32x16): lane<16 col N=lane holds K = k0..k0+15 contiguous,
// lane>=16 holds K = k0+16..k0+31 contiguous.
// ---------------------------------------------------------------------------
__global__ __launch_bounds__(256) void k_gemm_bf16_wmma(
    const bf16* __restrict__ A, const bf16* __restrict__ Bt,
    float* __restrict__ C, const float* __restrict__ bias,
    int M, int N, int K)
{
  const int wave = threadIdx.x >> 5;
  const int lane = threadIdx.x & 31;
  const int m0 = blockIdx.x * 128 + wave * 16;
  if (m0 >= M) return;                         // uniform per wave
  const int n0 = blockIdx.y * 64;              // N is a multiple of 64 here
  const int half = lane >> 4;
  const int r    = lane & 15;

  const bf16* arow = A  + (size_t)(m0 + r) * K;
  const bf16* brow0 = Bt + (size_t)(n0 +  0 + r) * K;
  const bf16* brow1 = Bt + (size_t)(n0 + 16 + r) * K;
  const bf16* brow2 = Bt + (size_t)(n0 + 32 + r) * K;
  const bf16* brow3 = Bt + (size_t)(n0 + 48 + r) * K;

  v8f acc0 = {}, acc1 = {}, acc2 = {}, acc3 = {};

  for (int k0 = 0; k0 < K; k0 += 32) {
    const int ka = k0 + half * 16;   // A: contiguous 16 halfs at ka
    const int kb = k0 + half * 16;   // B: contiguous 16 halfs at kb

    BFrag fa, f0, f1, f2, f3;
    fa.h[0] = *(const v8bf*)(arow + ka);
    fa.h[1] = *(const v8bf*)(arow + ka + 8);
    f0.h[0] = *(const v8bf*)(brow0 + kb);  f0.h[1] = *(const v8bf*)(brow0 + kb + 8);
    f1.h[0] = *(const v8bf*)(brow1 + kb);  f1.h[1] = *(const v8bf*)(brow1 + kb + 8);
    f2.h[0] = *(const v8bf*)(brow2 + kb);  f2.h[1] = *(const v8bf*)(brow2 + kb + 8);
    f3.h[0] = *(const v8bf*)(brow3 + kb);  f3.h[1] = *(const v8bf*)(brow3 + kb + 8);

    acc0 = __builtin_amdgcn_wmma_f32_16x16x32_bf16(false, fa.v, false, f0.v,
                                                   (short)0, acc0, false, false);
    acc1 = __builtin_amdgcn_wmma_f32_16x16x32_bf16(false, fa.v, false, f1.v,
                                                   (short)0, acc1, false, false);
    acc2 = __builtin_amdgcn_wmma_f32_16x16x32_bf16(false, fa.v, false, f2.v,
                                                   (short)0, acc2, false, false);
    acc3 = __builtin_amdgcn_wmma_f32_16x16x32_bf16(false, fa.v, false, f3.v,
                                                   (short)0, acc3, false, false);
  }

  // C/D layout: VGPR j -> row m0 + 8*half + j, col = n0 + 16*t + (lane&15)
  const int row = m0 + half * 8;
  float b0 = 0.f, b1 = 0.f, b2 = 0.f, b3 = 0.f;
  if (bias) {
    b0 = bias[n0 +  0 + r]; b1 = bias[n0 + 16 + r];
    b2 = bias[n0 + 32 + r]; b3 = bias[n0 + 48 + r];
  }
#pragma unroll
  for (int j = 0; j < 8; ++j) {
    size_t ro = (size_t)(row + j) * N;
    C[ro + n0 +  0 + r] = acc0[j] + b0;
    C[ro + n0 + 16 + r] = acc1[j] + b1;
    C[ro + n0 + 32 + r] = acc2[j] + b2;
    C[ro + n0 + 48 + r] = acc3[j] + b3;
  }
}

// ---------------------------------------------------------------------------
// Attention scores: es[n,h] = sum_c H[n,h,c]*a_src[h,c]; ed likewise.
// One wave per (node, head), C=512 -> 16 elems/lane + shfl_xor reduce.
// ---------------------------------------------------------------------------
__global__ void k_attn_scores(const float* __restrict__ H,
                              const float* __restrict__ asrc,
                              const float* __restrict__ adst,
                              float* __restrict__ es, float* __restrict__ ed,
                              int n, int heads, int C)
{
  int wid  = (int)((blockIdx.x * (size_t)blockDim.x + threadIdx.x) >> 5);
  int lane = threadIdx.x & 31;
  if (wid >= n * heads) return;
  int node = wid / heads;
  int h    = wid - node * heads;
  const float* hp = H + (size_t)node * heads * C + (size_t)h * C;
  const float* ap = asrc + (size_t)h * C;
  const float* dp = adst + (size_t)h * C;
  float s = 0.f, d = 0.f;
  for (int c = lane; c < C; c += 32) {
    float v = hp[c];
    s += v * ap[c];
    d += v * dp[c];
  }
#pragma unroll
  for (int off = 16; off > 0; off >>= 1) {
    s += __shfl_xor(s, off, 32);
    d += __shfl_xor(d, off, 32);
  }
  if (lane == 0) { es[wid] = s; ed[wid] = d; }
}

// ---------------------------------------------------------------------------
// Edge pass 1: e = leaky_relu(es[src]+ed[dst]); store; atomicMax emax[dst]
// ---------------------------------------------------------------------------
__global__ void k_edge_logits(const int* __restrict__ ei, int E0, int n, int heads,
                              const float* __restrict__ es, const float* __restrict__ ed,
                              float* __restrict__ alpha, unsigned* __restrict__ emax)
{
  int e = blockIdx.x * blockDim.x + threadIdx.x;
  int Et = E0 + n;
  if (e >= Et) return;
  int src = (e < E0) ? ei[e]      : (e - E0);
  int dst = (e < E0) ? ei[E0 + e] : (e - E0);
  for (int h = 0; h < heads; ++h) {
    float v = es[src * heads + h] + ed[dst * heads + h];
    v = (v > 0.f) ? v : 0.2f * v;                    // LeakyReLU(0.2)
    alpha[(size_t)e * heads + h] = v;
    atomicMax(&emax[dst * heads + h], f2ord(v));
  }
}

// ---------------------------------------------------------------------------
// Edge pass 2: a = exp(e - emax[dst]); store; atomicAdd den[dst]
// ---------------------------------------------------------------------------
__global__ void k_edge_softmax(const int* __restrict__ ei, int E0, int n, int heads,
                               const unsigned* __restrict__ emax,
                               float* __restrict__ alpha, float* __restrict__ den)
{
  int e = blockIdx.x * blockDim.x + threadIdx.x;
  int Et = E0 + n;
  if (e >= Et) return;
  int dst = (e < E0) ? ei[E0 + e] : (e - E0);
  for (int h = 0; h < heads; ++h) {
    float m = ord2f(emax[dst * heads + h]);          // self-loop guarantees valid
    float a = __expf(alpha[(size_t)e * heads + h] - m);
    alpha[(size_t)e * heads + h] = a;
    atomicAdd(&den[dst * heads + h], a);
  }
}

// ---------------------------------------------------------------------------
// Edge pass 3: out[dst, h, c] += H[src, h, c] * alpha[e,h] / den[dst,h]
// One block (256 threads) per edge; F = heads*C elements.
// ---------------------------------------------------------------------------
__global__ __launch_bounds__(256) void k_edge_aggregate(
    const int* __restrict__ ei, int E0, int n, int heads, int C,
    const float* __restrict__ H, const float* __restrict__ alpha,
    const float* __restrict__ den, float* __restrict__ out)
{
  int e = blockIdx.x;
  int src = (e < E0) ? ei[e]      : (e - E0);
  int dst = (e < E0) ? ei[E0 + e] : (e - E0);
  __shared__ float w[8];
  if ((int)threadIdx.x < heads)
    w[threadIdx.x] = alpha[(size_t)e * heads + threadIdx.x] /
                     den[dst * heads + threadIdx.x];
  __syncthreads();
  int F = heads * C;
  const float* hs = H   + (size_t)src * F;
  float*       od = out + (size_t)dst * F;
  for (int i = threadIdx.x; i < F; i += 256)
    atomicAdd(&od[i], hs[i] * w[i / C]);
}

// ---------------------------------------------------------------------------
// Epilogues
// ---------------------------------------------------------------------------
// next-layer input: bf16( elu(out + bias) )
__global__ void k_bias_elu_bf16(const float* __restrict__ out,
                                const float* __restrict__ bias,
                                bf16* __restrict__ A, size_t total, int F)
{
  size_t i = (size_t)blockIdx.x * blockDim.x + threadIdx.x;
  if (i >= total) return;
  float v = out[i] + bias[i % F];
  v = (v > 0.f) ? v : expm1f(v);                     // ELU(alpha=1)
  A[i] = f2bf(v);
}

// final layer: mean over 2 heads + bias -> d_out (f32)
__global__ void k_head_mean_bias(const float* __restrict__ out,
                                 const float* __restrict__ bias,
                                 float* __restrict__ y, int n, int C)
{
  size_t i = (size_t)blockIdx.x * blockDim.x + threadIdx.x;
  if (i >= (size_t)n * C) return;
  int node = (int)(i / C);
  int c    = (int)(i % C);
  y[i] = 0.5f * (out[(size_t)node * 2 * C + c] +
                 out[(size_t)node * 2 * C + C + c]) + bias[c];
}

// ---------------------------------------------------------------------------
// Host-side orchestration
// ---------------------------------------------------------------------------
extern "C" void kernel_launch(void* const* d_in, const int* in_sizes, int n_in,
                              void* d_out, int out_size, void* d_ws, size_t ws_size,
                              hipStream_t stream)
{
  const float* x      = (const float*)d_in[0];
  const int*   ei     = (const int*)  d_in[1];
  const float* proj_w = (const float*)d_in[2];
  const float* proj_b = (const float*)d_in[3];
  const float* Wl[3]  = {(const float*)d_in[4],  (const float*)d_in[8],  (const float*)d_in[12]};
  const float* ASl[3] = {(const float*)d_in[5],  (const float*)d_in[9],  (const float*)d_in[13]};
  const float* ADl[3] = {(const float*)d_in[6],  (const float*)d_in[10], (const float*)d_in[14]};
  const float* Bl[3]  = {(const float*)d_in[7],  (const float*)d_in[11], (const float*)d_in[15]};

  const int IN    = in_sizes[2] / in_sizes[3];   // 256
  const int n     = in_sizes[0] / IN;            // 10000
  const int E0    = in_sizes[1] / 2;             // 160000
  const int Et    = E0 + n;                      // with self-loops
  const int C     = 512;
  const int heads[3] = {4, 4, 2};
  const int Kdim[3]  = {512, 2048, 2048};
  const int Fdim[3]  = {2048, 2048, 1024};

  // ---- workspace carve-out ----
  char* ws = (char*)d_ws;
  size_t off = 0;
  auto align256 = [](size_t v) { return (v + 255) & ~(size_t)255; };
  bf16*     Abf   = (bf16*)(ws + off);     off = align256(off + (size_t)n * 2048 * sizeof(bf16));
  bf16*     Wt    = (bf16*)(ws + off);     off = align256(off + (size_t)2048 * 2048 * sizeof(bf16));
  float*    H     = (float*)(ws + off);    off = align256(off + (size_t)n * 2048 * sizeof(float));
  float*    OUT   = (float*)(ws + off);    off = align256(off + (size_t)n * 2048 * sizeof(float));
  float*    ES    = (float*)(ws + off);    off = align256(off + (size_t)n * 4 * sizeof(float));
  float*    ED    = (float*)(ws + off);    off = align256(off + (size_t)n * 4 * sizeof(float));
  unsigned* EMAX  = (unsigned*)(ws + off); off = align256(off + (size_t)n * 4 * sizeof(unsigned));
  float*    DEN   = (float*)(ws + off);    off = align256(off + (size_t)n * 4 * sizeof(float));
  float*    ALPHA = (float*)(ws + off);    off = align256(off + (size_t)Et * 4 * sizeof(float));

  const int T = 256;
  auto blks = [T](size_t total) { return (unsigned)((total + T - 1) / T); };

  // ---- projection: h0 = x @ proj_w + proj_b ----
  k_f32_to_bf16<<<blks((size_t)n * IN), T, 0, stream>>>(x, Abf, (size_t)n * IN);
  k_transpose_to_bf16<<<blks((size_t)IN * 512), T, 0, stream>>>(proj_w, Wt, IN, 512);
  {
    dim3 g((n + 127) / 128, 512 / 64);
    k_gemm_bf16_wmma<<<g, T, 0, stream>>>(Abf, Wt, H, proj_b, n, 512, IN);
  }
  k_f32_to_bf16<<<blks((size_t)n * 512), T, 0, stream>>>(H, Abf, (size_t)n * 512);

  // ---- 3 GAT layers ----
  for (int l = 0; l < 3; ++l) {
    const int hh = heads[l], K = Kdim[l], F = Fdim[l];

    // H = A @ W  (bf16 WMMA, f32 accum)
    k_transpose_to_bf16<<<blks((size_t)K * F), T, 0, stream>>>(Wl[l], Wt, K, F);
    {
      dim3 g((n + 127) / 128, F / 64);
      k_gemm_bf16_wmma<<<g, T, 0, stream>>>(Abf, Wt, H, nullptr, n, F, K);
    }

    // attention scores
    k_attn_scores<<<blks((size_t)n * hh * 32), T, 0, stream>>>(
        H, ASl[l], ADl[l], ES, ED, n, hh, C);

    // init segment buffers
    k_fill_u32<<<blks((size_t)n * hh), T, 0, stream>>>(EMAX, 0u, (size_t)n * hh);
    k_fill_f32<<<blks((size_t)n * hh), T, 0, stream>>>(DEN, 0.f, (size_t)n * hh);
    k_fill_f32<<<blks((size_t)n * F), T, 0, stream>>>(OUT, 0.f, (size_t)n * F);

    // segment softmax over dst + weighted aggregation
    k_edge_logits  <<<blks(Et), T, 0, stream>>>(ei, E0, n, hh, ES, ED, ALPHA, EMAX);
    k_edge_softmax <<<blks(Et), T, 0, stream>>>(ei, E0, n, hh, EMAX, ALPHA, DEN);
    k_edge_aggregate<<<Et, T, 0, stream>>>(ei, E0, n, hh, C, H, ALPHA, DEN, OUT);

    // epilogue
    if (l < 2) {
      k_bias_elu_bf16<<<blks((size_t)n * F), T, 0, stream>>>(
          OUT, Bl[l], Abf, (size_t)n * F, F);
    } else {
      k_head_mean_bias<<<blks((size_t)n * C), T, 0, stream>>>(
          OUT, Bl[2], (float*)d_out, n, C);
    }
  }
}
